// SelfAttention_50568944943649
// MI455X (gfx1250) — matrix operand order repaired and checked by reading, not compiled, not run on hardware
//
#include <hip/hip_runtime.h>
#include <hip/hip_bf16.h>

// ---------------------------------------------------------------------------
// Self-attention for B=4, S=2048, H=1024 (fp32 in/out), bf16 WMMA internally.
//   1) q/k/v projections:  [8192,1024] x [1024,1024]^T -> bf16
//   2) scores = q k^T / 32 per batch -> fp32
//   3) row softmax -> bf16 probs
//   4) out = probs @ v per batch -> fp32
// Matrix math: v_wmma_f32_16x16x32_bf16 (wave32, gfx1250).
// bf16 tiles: double-buffered LDS fed by global_load_async_to_lds_b128,
// overlapped with WMMA, synced via s_wait_asynccnt.
// ---------------------------------------------------------------------------

typedef __attribute__((ext_vector_type(16))) __bf16 v16bf;
typedef __attribute__((ext_vector_type(8)))  float  v8f;
typedef __attribute__((ext_vector_type(4)))  int    v4i;

#ifndef __has_builtin
#define __has_builtin(x) 0
#endif

#if __has_builtin(__builtin_amdgcn_global_load_async_to_lds_b128)
#define HAVE_ASYNC_LDS 1
#else
#define HAVE_ASYNC_LDS 0
#endif

#define AS_GLOBAL __attribute__((address_space(1)))
#define AS_SHARED __attribute__((address_space(3)))

__device__ inline void wait_async0() {
#if __has_builtin(__builtin_amdgcn_s_wait_asynccnt)
    __builtin_amdgcn_s_wait_asynccnt(0);
#else
    asm volatile("s_wait_asynccnt 0" ::: "memory");
#endif
}

__device__ inline unsigned short bf_bits(__bf16 b) {
    return __builtin_bit_cast(unsigned short, b);
}
__device__ inline void store_out(float* p, float v) { *p = v; }
__device__ inline void store_out(unsigned short* p, float v) {
    *p = bf_bits((__bf16)v);
}

// 16-byte copy global(bf16) -> LDS, async if the gfx1250 builtin exists.
__device__ inline void copy16_g2l(const unsigned short* gp, __bf16* lp) {
#if HAVE_ASYNC_LDS
    __builtin_amdgcn_global_load_async_to_lds_b128(
        (AS_GLOBAL v4i*)(unsigned short*)gp, (AS_SHARED v4i*)lp, 0, 0);
#else
    *(uint4*)lp = *(const uint4*)gp;
#endif
}

// Tiled GEMM:  C[M,N] = scale * A[M,K] * op(B)
//   BT=true : B stored as [N,K] row-major (computes A * B^T)
//   BT=false: B stored as [K,N] row-major (computes A * B)
// Block: 256 threads (8 waves). Tile 128x128, K-step 32, double-buffered LDS.
// Each wave: 64x32 region = 4(M) x 2(N) WMMA 16x16 tiles.
template <bool BT, typename TA, typename TB, typename TOUT>
__global__ __launch_bounds__(256)
void gemm_bf16_wmma(const TA* __restrict__ Ag, const TB* __restrict__ Bg,
                    TOUT* __restrict__ Cg,
                    int M, int N, int K, int lda, int ldb, int ldc,
                    long long strideA, long long strideB, long long strideC,
                    float scale) {
    constexpr int BM = 128, BN = 128, BK = 32;
    constexpr int LDA_S = BK + 8;                   // 40 elems = 80 B (16B mult)
    constexpr int LDB_S = BT ? (BK + 8) : (BN + 8); // 40 or 136 (272 B)
    constexpr int ASZ = BM * LDA_S;
    constexpr int BSZ = BT ? (BN * (BK + 8)) : (BK * (BN + 8));
    constexpr bool ASYNC =
        HAVE_ASYNC_LDS && (sizeof(TA) == 2 || sizeof(TB) == 2);

    __shared__ __bf16 ldsA[2 * ASZ];
    __shared__ __bf16 ldsB[2 * BSZ];

    const TA* A  = Ag + (long long)blockIdx.z * strideA;
    const TB* Bm = Bg + (long long)blockIdx.z * strideB;
    TOUT*     C  = Cg + (long long)blockIdx.z * strideC;

    const int tid  = threadIdx.x;
    const int lane = tid & 31;
    const int wave = tid >> 5;
    const int m0 = blockIdx.y * BM;
    const int n0 = blockIdx.x * BN;
    const int wm = (wave >> 2) * 64;   // wave row offset in tile: 0 or 64
    const int wn = (wave & 3) * 32;    // wave col offset in tile: 0..96

    // ---- tile staging (issue loads first, then convert/store) ----
    auto stageA = [&](int buf, int k0) {
        __bf16* dst = ldsA + buf * ASZ;
        if constexpr (sizeof(TA) == 4) {
            constexpr int IT = (BM * BK / 4) / 256;  // 4
            float4 f[IT];
#pragma unroll
            for (int it = 0; it < IT; ++it) {
                const int e4 = tid + it * 256;
                const int m = e4 >> 3, k = (e4 & 7) * 4;
                f[it] = *(const float4*)&A[(long long)(m0 + m) * lda + k0 + k];
            }
#pragma unroll
            for (int it = 0; it < IT; ++it) {
                const int e4 = tid + it * 256;
                const int m = e4 >> 3, k = (e4 & 7) * 4;
                union { unsigned short s[4]; uint2 v; } u;
                u.s[0] = bf_bits((__bf16)f[it].x);
                u.s[1] = bf_bits((__bf16)f[it].y);
                u.s[2] = bf_bits((__bf16)f[it].z);
                u.s[3] = bf_bits((__bf16)f[it].w);
                *(uint2*)&dst[m * LDA_S + k] = u.v;
            }
        } else {
#pragma unroll
            for (int it = 0; it < (BM * BK / 8) / 256; ++it) {
                const int e8 = tid + it * 256;
                const int m = e8 >> 2, k = (e8 & 3) * 8;
                copy16_g2l((const unsigned short*)A +
                               (long long)(m0 + m) * lda + k0 + k,
                           &dst[m * LDA_S + k]);
            }
        }
    };
    auto stageB = [&](int buf, int k0) {
        __bf16* dst = ldsB + buf * BSZ;
        if constexpr (BT) {
            if constexpr (sizeof(TB) == 4) {
                constexpr int IT = (BN * BK / 4) / 256;  // 4
                float4 f[IT];
#pragma unroll
                for (int it = 0; it < IT; ++it) {
                    const int e4 = tid + it * 256;
                    const int n = e4 >> 3, k = (e4 & 7) * 4;
                    f[it] = *(const float4*)&Bm[(long long)(n0 + n) * ldb + k0 +
                                                k];
                }
#pragma unroll
                for (int it = 0; it < IT; ++it) {
                    const int e4 = tid + it * 256;
                    const int n = e4 >> 3, k = (e4 & 7) * 4;
                    union { unsigned short s[4]; uint2 v; } u;
                    u.s[0] = bf_bits((__bf16)f[it].x);
                    u.s[1] = bf_bits((__bf16)f[it].y);
                    u.s[2] = bf_bits((__bf16)f[it].z);
                    u.s[3] = bf_bits((__bf16)f[it].w);
                    *(uint2*)&dst[n * LDB_S + k] = u.v;
                }
            } else {
#pragma unroll
                for (int it = 0; it < (BN * BK / 8) / 256; ++it) {
                    const int e8 = tid + it * 256;
                    const int n = e8 >> 2, k = (e8 & 3) * 8;
                    copy16_g2l((const unsigned short*)Bm +
                                   (long long)(n0 + n) * ldb + k0 + k,
                               &dst[n * LDB_S + k]);
                }
            }
        } else {
            // [32 x 128] bf16 tile, 16 chunks of 8 per row
#pragma unroll
            for (int it = 0; it < (BK * BN / 8) / 256; ++it) {
                const int e8 = tid + it * 256;
                const int k = e8 >> 4, n = (e8 & 15) * 8;
                copy16_g2l((const unsigned short*)Bm +
                               (long long)(k0 + k) * ldb + n0 + n,
                           &dst[k * LDB_S + n]);
            }
        }
    };

    const v8f vzero = (v8f)(0.0f);
    v8f acc[4][2];
#pragma unroll
    for (int i = 0; i < 4; ++i)
#pragma unroll
        for (int j = 0; j < 2; ++j) acc[i][j] = vzero;

    // ---- prologue: stage first K-tile into buffer 0 ----
    stageA(0, 0);
    stageB(0, 0);
    if constexpr (ASYNC) wait_async0();
    __syncthreads();

    int cur = 0;
    for (int k0 = 0; k0 < K; k0 += BK) {
        const int nxt = cur ^ 1;
        const bool more = (k0 + BK) < K;
        // issue next tile's (async) loads before computing on current tile
        if (more) {
            stageA(nxt, k0 + BK);
            stageB(nxt, k0 + BK);
        }
        const __bf16* As = ldsA + cur * ASZ;
        const __bf16* Bs = ldsB + cur * BSZ;

        // ---- A fragments (16x32 bf16 layout, wave32) ----
        // lane 0-15: M=lane, K pairs {0..7,16..23}; lane 16-31: +8 on K pairs
        const int mrow = lane & 15;
        const int hiA  = (lane >> 4) * 8;
        v16bf afr[4];
#pragma unroll
        for (int t = 0; t < 4; ++t) {
            const int mbase = wm + t * 16 + mrow;
#pragma unroll
            for (int r = 0; r < 8; ++r) {
                const int kp = ((r < 4) ? (2 * r) : (16 + 2 * (r - 4))) + hiA;
                afr[t][2 * r]     = As[mbase * LDA_S + kp];
                afr[t][2 * r + 1] = As[mbase * LDA_S + kp + 1];
            }
        }
        // ---- B fragments (32x16 bf16): N=lane%16, K = (j<8 ? j : 8+j) + (lane/16)*8 ----
        const int nlane = lane & 15;
        const int kHi   = (lane >> 4) * 8;
        v16bf bfr[2];
#pragma unroll
        for (int t = 0; t < 2; ++t) {
            const int nb = wn + t * 16 + nlane;
#pragma unroll
            for (int j = 0; j < 16; ++j) {
                const int kk = ((j < 8) ? j : (8 + j)) + kHi;
                if constexpr (BT)
                    bfr[t][j] = Bs[nb * LDB_S + kk];
                else
                    bfr[t][j] = Bs[kk * LDB_S + nb];
            }
        }
        // ---- 8 WMMAs per wave per K-step (overlaps with async DMA) ----
#pragma unroll
        for (int i = 0; i < 4; ++i)
#pragma unroll
            for (int j = 0; j < 2; ++j)
                acc[i][j] = __builtin_amdgcn_wmma_f32_16x16x32_bf16(
                    false, afr[i], false, bfr[j], (short)0, acc[i][j], false,
                    false);

        if (more) {
            if constexpr (ASYNC) wait_async0();
        }
        __syncthreads();
        cur = nxt;
    }

    // ---- store C: lane holds N=lane%16; VGPR r -> M = r + 8*(lane/16) ----
    const int nlane = lane & 15;
    const int mOff  = (lane >> 4) * 8;
#pragma unroll
    for (int i = 0; i < 4; ++i)
#pragma unroll
        for (int j = 0; j < 2; ++j)
#pragma unroll
            for (int r = 0; r < 8; ++r) {
                const int m = m0 + wm + i * 16 + mOff + r;
                const int n = n0 + wn + j * 16 + nlane;
                store_out(&C[(long long)m * ldc + n], acc[i][j][r] * scale);
            }
}

// Row softmax: one 256-thread block per row of 2048 fp32 -> bf16 probs.
__global__ __launch_bounds__(256)
void softmax_rows_bf16(const float* __restrict__ Sc,
                       unsigned short* __restrict__ P, int ncols) {
    const long long row = blockIdx.x;
    const float* src = Sc + row * ncols;
    unsigned short* dst = P + row * ncols;
    const int tid = threadIdx.x;

    float v[8];
#pragma unroll
    for (int i = 0; i < 8; ++i) v[i] = src[tid + i * 256];

    __shared__ float red[8];
    __shared__ float bmax, bsum;

    float m = -3.0e38f;
#pragma unroll
    for (int i = 0; i < 8; ++i) m = fmaxf(m, v[i]);
#pragma unroll
    for (int off = 16; off > 0; off >>= 1) m = fmaxf(m, __shfl_xor(m, off, 32));
    if ((tid & 31) == 0) red[tid >> 5] = m;
    __syncthreads();
    if (tid == 0) {
        float t = red[0];
        for (int i = 1; i < 8; ++i) t = fmaxf(t, red[i]);
        bmax = t;
    }
    __syncthreads();
    m = bmax;

    float s = 0.0f;
#pragma unroll
    for (int i = 0; i < 8; ++i) {
        v[i] = __expf(v[i] - m);
        s += v[i];
    }
#pragma unroll
    for (int off = 16; off > 0; off >>= 1) s += __shfl_xor(s, off, 32);
    if ((tid & 31) == 0) red[tid >> 5] = s;
    __syncthreads();
    if (tid == 0) {
        float t = 0.0f;
        for (int i = 0; i < 8; ++i) t += red[i];
        bsum = t;
    }
    __syncthreads();
    const float inv = 1.0f / bsum;
#pragma unroll
    for (int i = 0; i < 8; ++i)
        dst[tid + i * 256] = bf_bits((__bf16)(v[i] * inv));
}

extern "C" void kernel_launch(void* const* d_in, const int* in_sizes, int n_in,
                              void* d_out, int out_size, void* d_ws,
                              size_t ws_size, hipStream_t stream) {
    (void)in_sizes; (void)n_in; (void)out_size; (void)ws_size;

    const float* query = (const float*)d_in[0];
    const float* key_  = (const float*)d_in[1];
    const float* value = (const float*)d_in[2];
    const float* Wq    = (const float*)d_in[3];
    const float* Wk    = (const float*)d_in[4];
    const float* Wv    = (const float*)d_in[5];
    float* out = (float*)d_out;

    constexpr int Bb = 4, S = 2048, H = 1024;
    constexpr long long M = (long long)Bb * S;  // 8192

    // workspace layout (bf16 stored as u16):
    //   qb,kb,vb: [B*S, H] bf16 each; probs: [B, S, S] bf16; scores: fp32
    unsigned short* qb = (unsigned short*)d_ws;
    unsigned short* kb = qb + M * H;
    unsigned short* vb = kb + M * H;
    unsigned short* pb = vb + M * H;
    float* scores = (float*)(pb + (long long)Bb * S * S);

    dim3 blk(256);

    // 1) projections: Y = X * W^T  (M=8192, N=1024, K=1024)
    dim3 gproj(H / 128, (int)(M / 128), 1);
    auto* proj = &gemm_bf16_wmma<true, float, float, unsigned short>;
    hipLaunchKernelGGL(proj, gproj, blk, 0, stream, query, Wq, qb,
                       (int)M, H, H, H, H, H, 0LL, 0LL, 0LL, 1.0f);
    hipLaunchKernelGGL(proj, gproj, blk, 0, stream, key_, Wk, kb,
                       (int)M, H, H, H, H, H, 0LL, 0LL, 0LL, 1.0f);
    hipLaunchKernelGGL(proj, gproj, blk, 0, stream, value, Wv, vb,
                       (int)M, H, H, H, H, H, 0LL, 0LL, 0LL, 1.0f);

    // 2) scores = q k^T / sqrt(H), per batch (grid.z = B)
    dim3 gsc(S / 128, S / 128, Bb);
    auto* scgemm =
        &gemm_bf16_wmma<true, unsigned short, unsigned short, float>;
    hipLaunchKernelGGL(scgemm, gsc, blk, 0, stream, qb, kb, scores,
                       S, S, H, H, H, S,
                       (long long)S * H, (long long)S * H, (long long)S * S,
                       0.03125f);

    // 3) row softmax -> bf16 probs
    hipLaunchKernelGGL(softmax_rows_bf16, dim3((unsigned)(Bb * S)), blk, 0,
                       stream, scores, pb, S);

    // 4) out = probs @ v, per batch
    dim3 gout(H / 128, S / 128, Bb);
    auto* ogemm =
        &gemm_bf16_wmma<false, unsigned short, unsigned short, float>;
    hipLaunchKernelGGL(ogemm, gout, blk, 0, stream, pb, vb, out,
                       S, H, S, S, H, H,
                       (long long)S * S, (long long)S * H, (long long)S * H,
                       1.0f);
}
